// Net_56100862820850
// MI455X (gfx1250) — compile-verified, matrix operand order checked by reading, not run on hardware
//
#include <hip/hip_runtime.h>
#include <hip/hip_bf16.h>

// ---------------- problem constants (from reference) ----------------
#define D_DIM   128
#define S_DIM   255          // 2D-1 shift candidates
#define XPAD    382          // D + 2*(D-1)
#define HDIM    1024
#define CDIM    128
#define BT      4096         // B*T = 8*512
#define NITER   8
#define INV_TEMPER (1.0f/0.85f)
#define ENERGY_TH 0.01f

typedef __bf16 bf16;
typedef bf16         v16bf __attribute__((ext_vector_type(16)));
typedef float        v8f   __attribute__((ext_vector_type(8)));
typedef unsigned int v8u   __attribute__((ext_vector_type(8)));

// scalars[] layout: 0=loss_total 1=diff_sum 2=valid_cnt 3=loss_h
#define S_TOTAL 0
#define S_DIFF  1
#define S_VALID 2
#define S_LOSSH 3

// K-offset inside a 32-wide k-block for fragment half i, lane-half hi
// (per CDNA5 ISA 16-bit A/B VGPR layout)
__device__ __forceinline__ int frag_koff(int hi, int i) {
  return (i < 8) ? (hi * 8 + i) : (16 + hi * 8 + (i - 8));
}

// ---------------- init ----------------
__global__ void k_init(const float* __restrict__ x, const float* __restrict__ y,
                       float* __restrict__ x_res, float* __restrict__ y_res,
                       float* __restrict__ mask_prev, float* __restrict__ scalars) {
  int i = blockIdx.x * blockDim.x + threadIdx.x;
  if (i < BT * D_DIM) { x_res[i] = x[i]; y_res[i] = y[i]; }
  if (i < HDIM) mask_prev[i] = 0.f;
  if (i < 8) scalars[i] = 0.f;
}

// swizzle a K x N row-major f32 weight matrix into per-wave B-fragment order:
// dst[(((tn*(K/32) + kb)*32 + lane)*16 + i] = (bf16)src[k*N + n]
__global__ void k_swizzle_w(const float* __restrict__ src, bf16* __restrict__ dst,
                            int K, int N) {
  int idx = blockIdx.x * blockDim.x + threadIdx.x;
  if (idx >= K * N) return;
  int i    = idx & 15;
  int lane = (idx >> 4) & 31;
  int t    = idx >> 9;
  int KB   = K >> 5;
  int kb   = t % KB;
  int tn   = t / KB;
  int hi   = lane >> 4, mrow = lane & 15;
  int k    = kb * 32 + frag_koff(hi, i);
  int n    = tn * 16 + mrow;
  dst[idx] = (bf16)src[(size_t)k * N + n];
}

__global__ void k_zero_iter(float* __restrict__ energy, float* __restrict__ scalars) {
  int i = blockIdx.x * blockDim.x + threadIdx.x;
  if (i < HDIM) energy[i] = 0.f;
  if (i == 0) { scalars[S_DIFF] = 0.f; scalars[S_VALID] = 0.f; scalars[S_LOSSH] = 0.f; }
}

// ---------------- selector + attention + x_res update ----------------
// one block per (b,t) row, 256 threads; writes encode-A in swizzled fragment order
__global__ void k_selector(const float* __restrict__ y_res, float* __restrict__ x_res,
                           bf16* __restrict__ Asw) {
  __shared__ float xp[XPAD];
  __shared__ float yv[D_DIM];
  __shared__ float sv[256];
  __shared__ int   si[256];
  __shared__ float ya[D_DIM];
  __shared__ float red[128];
  __shared__ float s_ny;

  const int row = blockIdx.x;
  const int tid = threadIdx.x;
  const float* xr = x_res + (size_t)row * D_DIM;

  for (int i = tid; i < XPAD; i += 256) {
    int d = i - (D_DIM - 1);
    xp[i] = (d >= 0 && d < D_DIM) ? xr[d] : 0.f;
  }
  if (tid < D_DIM) yv[tid] = y_res[(size_t)row * D_DIM + tid];
  __syncthreads();

  // ||y||
  if (tid < 128) red[tid] = yv[tid] * yv[tid];
  __syncthreads();
  for (int o = 64; o > 0; o >>= 1) { if (tid < o) red[tid] += red[tid + o]; __syncthreads(); }
  if (tid == 0) s_ny = sqrtf(red[0]);
  __syncthreads();
  const float ny = s_ny;
  __syncthreads();

  // per-shift cosine similarity
  float simv = -__builtin_inff();
  if (tid < S_DIM) {
    float nu = 0.f, nx = 0.f;
    #pragma unroll 4
    for (int d = 0; d < D_DIM; ++d) {
      float xw = xp[tid + d];
      nu += xw * yv[d];
      nx += xw * xw;
    }
    float denom = sqrtf(nx) * ny;
    simv = (denom > 0.f) ? (nu / fmaxf(denom, 1e-12f)) : 0.f;
  }
  sv[tid] = simv; si[tid] = tid;
  __syncthreads();
  for (int o = 128; o > 0; o >>= 1) {        // argmax, first index wins ties
    if (tid < o) {
      float v2 = sv[tid + o]; int i2 = si[tid + o];
      if (v2 > sv[tid] || (v2 == sv[tid] && i2 < si[tid])) { sv[tid] = v2; si[tid] = i2; }
    }
    __syncthreads();
  }
  const int theta = si[0];
  __syncthreads();

  // softmax(y_align * y / T)
  float yal = 0.f, logit = -__builtin_inff();
  if (tid < D_DIM) {
    yal = xp[theta + tid];
    logit = yal * yv[tid] * INV_TEMPER;
    red[tid] = logit;
  }
  __syncthreads();
  for (int o = 64; o > 0; o >>= 1) { if (tid < o) red[tid] = fmaxf(red[tid], red[tid + o]); __syncthreads(); }
  const float mx = red[0];
  __syncthreads();
  float e = 0.f;
  if (tid < D_DIM) { e = __expf(logit - mx); red[tid] = e; }
  __syncthreads();
  for (int o = 64; o > 0; o >>= 1) { if (tid < o) red[tid] += red[tid + o]; __syncthreads(); }
  const float ssum = red[0];
  if (tid < D_DIM) {
    float v = yal * (e / ssum);
    ya[tid] = v;
    // scatter into swizzled encode-A fragment layout (K = D_DIM, KB = 4)
    int kb = tid >> 5, r = tid & 31;
    int hi = (r >> 3) & 1;
    int i  = (r & 7) + ((r & 16) ? 8 : 0);
    int lane = (row & 15) + 16 * hi;
    int tm = row >> 4;
    Asw[(((size_t)tm * 4 + kb) * 32 + lane) * 16 + i] = (bf16)v;
  }
  __syncthreads();

  // x_res -= reverse_pad_for_shift(y_att, theta)
  if (tid < D_DIM) {
    int j = tid + (D_DIM - 1) - theta;
    float xe = (j >= 0 && j < D_DIM) ? ya[j] : 0.f;
    x_res[(size_t)row * D_DIM + tid] = xr[tid] - xe;
  }
}

// ---------------- encode: h = relu(Asw @ W_enc) -> Hsw (swizzled bf16), energy ----------------
// Asw: fragment-ordered bf16 (KB=4); Bsw: fragment-ordered W_enc; one wave per 16x16 tile.
__global__ void k_encode(const bf16* __restrict__ Asw, const bf16* __restrict__ Bsw,
                         bf16* __restrict__ Hsw, float* __restrict__ energy) {
  const int wave = (blockIdx.x * blockDim.x + threadIdx.x) >> 5;
  const int lane = threadIdx.x & 31;
  const int NT = HDIM / 16;                 // 64 col tiles
  const int tn = wave % NT, tm = wave / NT;
  const int mrow = lane & 15, hi = lane >> 4;

  v8f acc;
  #pragma unroll
  for (int r = 0; r < 8; ++r) acc[r] = 0.f;

  const v16bf* ap = (const v16bf*)(Asw + ((size_t)tm * 4 * 32 + lane) * 16);
  const v16bf* bp = (const v16bf*)(Bsw + ((size_t)tn * 4 * 32 + lane) * 16);
  #pragma unroll
  for (int kb = 0; kb < 4; ++kb) {
    v16bf a = ap[kb * 32];                  // 32B contiguous per lane
    v16bf b = bp[kb * 32];
    acc = __builtin_amdgcn_wmma_f32_16x16x32_bf16(false, a, false, b, (short)0, acc, false, false);
  }

  // relu, energy, and store h directly in decode-A swizzled fragment order (KB2 = 32)
  const int n   = tn * 16 + mrow;
  const int kb2 = n >> 5, r2 = n & 31;
  const int hi2 = (r2 >> 3) & 1;
  const int i2  = (r2 & 7) + ((r2 & 16) ? 8 : 0);
  float esum = 0.f;
  #pragma unroll
  for (int r = 0; r < 8; ++r) {
    float v = acc[r];
    v = v > 0.f ? v : 0.f;                  // relu
    int mrow2 = r + 8 * hi;                 // row within the m-tile
    int lane2 = mrow2 + 16 * hi2;
    Hsw[(((size_t)tm * 32 + kb2) * 32 + lane2) * 16 + i2] = (bf16)v;
    esum += v * v;                          // all rows of a lane share column n
  }
  atomicAdd(&energy[n], esum);
}

// ---------------- top-k selection + loss_h + fragment AND-mask ----------------
__global__ void k_topk(const float* __restrict__ energy, float* __restrict__ mask_prev,
                       unsigned short* __restrict__ mask_frag, float* __restrict__ scalars) {
  __shared__ float avail[HDIM];
  __shared__ float sel[HDIM];
  __shared__ float rv[256];
  __shared__ int   ri[256];
  const int tid = threadIdx.x;
  const float invBT = 1.f / (float)BT;

  for (int j = tid; j < HDIM; j += 256) {
    avail[j] = (mask_prev[j] > 0.f) ? -__builtin_inff() : energy[j] * invBT;
    sel[j] = 0.f;
  }
  __syncthreads();

  for (int it = 0; it < CDIM; ++it) {
    float bv = -__builtin_inff(); int bi = HDIM;
    for (int j = tid; j < HDIM; j += 256) {
      float v = avail[j];
      if (v > bv || (v == bv && j < bi)) { bv = v; bi = j; }
    }
    rv[tid] = bv; ri[tid] = bi;
    __syncthreads();
    for (int o = 128; o > 0; o >>= 1) {
      if (tid < o) {
        if (rv[tid + o] > rv[tid] || (rv[tid + o] == rv[tid] && ri[tid + o] < ri[tid])) {
          rv[tid] = rv[tid + o]; ri[tid] = ri[tid + o];
        }
      }
      __syncthreads();
    }
    if (tid == 0) { int k = ri[0]; sel[k] = 1.f; avail[k] = -__builtin_inff(); }
    __syncthreads();
  }

  // loss_h = energy of unselected units / (BT*HDIM); update mask_prev
  float lh = 0.f;
  for (int j = tid; j < HDIM; j += 256) {
    if (sel[j] > 0.f) mask_prev[j] += 1.f;
    lh += energy[j] * (1.f - sel[j]);
  }
  rv[tid] = lh;
  __syncthreads();
  for (int o = 128; o > 0; o >>= 1) { if (tid < o) rv[tid] += rv[tid + o]; __syncthreads(); }
  if (tid == 0) scalars[S_LOSSH] = rv[0] / ((float)BT * (float)HDIM);

  // 32-byte AND masks in fragment order: half-word 0xFFFF for selected k, 0 otherwise
  if (tid < (HDIM / 32) * 2) {
    int kb = tid >> 1, hi = tid & 1;
    #pragma unroll
    for (int i = 0; i < 16; ++i)
      mask_frag[tid * 16 + i] =
          (sel[kb * 32 + frag_koff(hi, i)] > 0.f) ? (unsigned short)0xFFFFu : (unsigned short)0u;
  }
}

// ---------------- decode: y_ele = (h * mask) @ W_dec ----------------
// Hsw: fragment-ordered bf16 (KB=32); mask applied as bitwise AND with 32B fragment mask.
__global__ void k_decode(const bf16* __restrict__ Hsw, const unsigned short* __restrict__ mask_frag,
                         const bf16* __restrict__ Bsw, float* __restrict__ y_ele) {
  const int wave = (blockIdx.x * blockDim.x + threadIdx.x) >> 5;
  const int lane = threadIdx.x & 31;
  const int NT = D_DIM / 16;                // 8 col tiles
  const int tn = wave % NT, tm = wave / NT;
  const int m0 = tm * 16, n0 = tn * 16;
  const int mrow = lane & 15, hi = lane >> 4;

  v8f acc;
  #pragma unroll
  for (int r = 0; r < 8; ++r) acc[r] = 0.f;

  const v8u* ap = (const v8u*)(Hsw + ((size_t)tm * 32 * 32 + lane) * 16);
  const v8u* mp = (const v8u*)(mask_frag + hi * 16);       // stride 2 v8u per k-block
  const v16bf* bp = (const v16bf*)(Bsw + ((size_t)tn * 32 * 32 + lane) * 16);

  #pragma unroll 4
  for (int kb = 0; kb < 32; ++kb) {
    if (kb + 1 < 32) {                      // prefetch next k-block fragments
      __builtin_prefetch((const void*)(ap + (kb + 1) * 32), 0, 1);
      __builtin_prefetch((const void*)(bp + (kb + 1) * 32), 0, 1);
    }
    v8u au = ap[kb * 32] & mp[kb * 2];      // gate h lanes with 8 dword ANDs
    v16bf a = __builtin_bit_cast(v16bf, au);
    v16bf b = bp[kb * 32];
    acc = __builtin_amdgcn_wmma_f32_16x16x32_bf16(false, a, false, b, (short)0, acc, false, false);
  }
  #pragma unroll
  for (int r = 0; r < 8; ++r)
    y_ele[(size_t)(m0 + r + 8 * hi) * D_DIM + n0 + mrow] = acc[r];
}

// ---------------- per-row seq_energy_loss + y_res update ----------------
__global__ void k_rowloss(const float* __restrict__ y_in, const float* __restrict__ y_ele,
                          float* __restrict__ y_res, float* __restrict__ scalars) {
  __shared__ float red[128];
  __shared__ float red2[128];
  const int row = blockIdx.x, tid = threadIdx.x;
  const size_t off = (size_t)row * D_DIM + tid;
  const float yr = y_res[off], ye = y_ele[off], yo = y_in[off];

  red[tid] = yr * yr;
  __syncthreads();
  for (int o = 64; o > 0; o >>= 1) { if (tid < o) red[tid] += red[tid + o]; __syncthreads(); }
  const float energy_t = red[0] * (1.f / (float)D_DIM);
  __syncthreads();

  const bool valid = (yo != 0.f) && (energy_t > ENERGY_TH);
  const float d = ye - yr;
  red[tid]  = valid ? d * d : 0.f;
  red2[tid] = valid ? 1.f : 0.f;
  __syncthreads();
  for (int o = 64; o > 0; o >>= 1) {
    if (tid < o) { red[tid] += red[tid + o]; red2[tid] += red2[tid + o]; }
    __syncthreads();
  }
  if (tid == 0) { atomicAdd(&scalars[S_DIFF], red[0]); atomicAdd(&scalars[S_VALID], red2[0]); }
  y_res[off] = yr - ye;
}

__global__ void k_iter_fin(float* __restrict__ scalars) {
  float vc = scalars[S_VALID];
  float denom = vc > 1.f ? vc : 1.f;
  scalars[S_TOTAL] += scalars[S_DIFF] / denom + scalars[S_LOSSH];
}

__global__ void k_final(const float* __restrict__ scalars, float* __restrict__ out) {
  out[0] = scalars[S_TOTAL] * (1.f / (float)NITER);
}

// ---------------- host-side pipeline ----------------
extern "C" void kernel_launch(void* const* d_in, const int* in_sizes, int n_in,
                              void* d_out, int out_size, void* d_ws, size_t ws_size,
                              hipStream_t stream) {
  const float* x  = (const float*)d_in[0];
  const float* y  = (const float*)d_in[1];
  const float* We = (const float*)d_in[2];
  const float* Wd = (const float*)d_in[3];

  char* ws = (char*)d_ws;
  size_t off = 0;
  auto alloc = [&](size_t bytes) -> void* {
    void* p = ws + off;
    off = (off + bytes + 255) & ~(size_t)255;
    return p;
  };

  float*          x_res     = (float*)alloc((size_t)BT * D_DIM * 4);
  float*          y_res     = (float*)alloc((size_t)BT * D_DIM * 4);
  float*          y_ele     = (float*)alloc((size_t)BT * D_DIM * 4);
  bf16*           Asw       = (bf16*)alloc((size_t)BT * D_DIM * 2);      // encode A, swizzled
  bf16*           Hsw       = (bf16*)alloc((size_t)BT * HDIM * 2);       // decode A, swizzled
  bf16*           BswE      = (bf16*)alloc((size_t)D_DIM * HDIM * 2);    // W_enc, swizzled
  bf16*           BswD      = (bf16*)alloc((size_t)HDIM * D_DIM * 2);    // W_dec, swizzled
  float*          energy    = (float*)alloc((size_t)HDIM * 4);
  float*          mask_prev = (float*)alloc((size_t)HDIM * 4);
  unsigned short* mask_frag = (unsigned short*)alloc((size_t)(HDIM / 32) * 2 * 16 * 2);
  float*          scalars   = (float*)alloc(256);

  k_init<<<(BT * D_DIM + 255) / 256, 256, 0, stream>>>(x, y, x_res, y_res, mask_prev, scalars);
  k_swizzle_w<<<(D_DIM * HDIM + 255) / 256, 256, 0, stream>>>(We, BswE, D_DIM, HDIM);
  k_swizzle_w<<<(HDIM * D_DIM + 255) / 256, 256, 0, stream>>>(Wd, BswD, HDIM, D_DIM);

  for (int it = 0; it < NITER; ++it) {
    k_selector<<<BT, 256, 0, stream>>>(y_res, x_res, Asw);
    k_zero_iter<<<(HDIM + 255) / 256, 256, 0, stream>>>(energy, scalars);
    // encode: (BT/16)*(HDIM/16) = 16384 wave-tiles, 8 waves per 256-thread block
    k_encode<<<(BT / 16) * (HDIM / 16) / 8, 256, 0, stream>>>(Asw, BswE, Hsw, energy);
    k_topk<<<1, 256, 0, stream>>>(energy, mask_prev, mask_frag, scalars);
    // decode: (BT/16)*(D/16) = 2048 wave-tiles
    k_decode<<<(BT / 16) * (D_DIM / 16) / 8, 256, 0, stream>>>(Hsw, mask_frag, BswD, y_ele);
    k_rowloss<<<BT, 128, 0, stream>>>(y, y_ele, y_res, scalars);
    k_iter_fin<<<1, 1, 0, stream>>>(scalars);
  }
  k_final<<<1, 1, 0, stream>>>(scalars, (float*)d_out);
}
